// PVDST_semseg_11673721111019
// MI455X (gfx1250) — matrix operand order repaired and checked by reading, not compiled, not run on hardware
//
#include <hip/hip_runtime.h>
#include <hip/hip_bf16.h>

typedef __attribute__((ext_vector_type(16))) _Float16 v16h;
typedef __attribute__((ext_vector_type(8)))  float    v8f;

#define BB 4
#define NN 4096
#define CC 128

// ---------------------------------------------------------------------------
// f32 -> f16 convert (contiguous)
// ---------------------------------------------------------------------------
__global__ void cvt_f16_kernel(const float* __restrict__ src, _Float16* __restrict__ dst, int n) {
    int i = blockIdx.x * 256 + threadIdx.x;
    if (i < n) dst[i] = (_Float16)src[i];
}

// cls_w1[:, :1024] submatrix convert (512 x 3072 source)
__global__ void cvt_w1a_kernel(const float* __restrict__ src, _Float16* __restrict__ dst) {
    int i = blockIdx.x * 256 + threadIdx.x;    // 512*1024
    int o = i >> 10, c = i & 1023;
    dst[i] = (_Float16)src[(size_t)o * 3072 + c];
}

// ---------------------------------------------------------------------------
// kNN: one wave per query point. Lane-local sorted top-16 over strided
// candidates, then 16-round 32-lane merge via shfl_xor (wave32).
// ---------------------------------------------------------------------------
__global__ void knn_kernel(const float* __restrict__ in, int* __restrict__ idx) {
    const int lane = threadIdx.x & 31;
    const int wv   = threadIdx.x >> 5;
    const int p    = blockIdx.x * 8 + wv;          // B*N points
    const int b = p / NN, n = p % NN;
    const float* xb = in + (size_t)b * 9 * NN;
    const float qx = xb[n], qy = xb[NN + n], qz = xb[2 * NN + n];
    const float sqn = qx * qx + qy * qy + qz * qz;

    float dl[16]; int il[16];
#pragma unroll
    for (int t = 0; t < 16; ++t) { dl[t] = 3.4e38f; il[t] = 0x7fffffff; }

    for (int m = lane; m < NN; m += 32) {
        float mx = xb[m], my = xb[NN + m], mz = xb[2 * NN + m];
        float d = sqn + (mx * mx + my * my + mz * mz) - 2.f * (qx * mx + qy * my + qz * mz);
        if (d < dl[15] || (d == dl[15] && m < il[15])) {
            dl[15] = d; il[15] = m;
#pragma unroll
            for (int t = 15; t > 0; --t) {
                bool sw = (dl[t] < dl[t - 1]) || (dl[t] == dl[t - 1] && il[t] < il[t - 1]);
                if (sw) {
                    float td = dl[t]; dl[t] = dl[t - 1]; dl[t - 1] = td;
                    int   ti = il[t]; il[t] = il[t - 1]; il[t - 1] = ti;
                }
            }
        }
    }

    int* op = idx + ((size_t)b * NN + n) * 16;
    for (int k = 0; k < 16; ++k) {
        float dbest = 3.4e38f; int ibest = 0x7fffffff;
#pragma unroll
        for (int t = 0; t < 16; ++t)
            if (dl[t] < dbest || (dl[t] == dbest && il[t] < ibest)) { dbest = dl[t]; ibest = il[t]; }
#pragma unroll
        for (int off = 16; off; off >>= 1) {
            float od = __shfl_xor(dbest, off, 32);
            int   oi = __shfl_xor(ibest, off, 32);
            if (od < dbest || (od == dbest && oi < ibest)) { dbest = od; ibest = oi; }
        }
#pragma unroll
        for (int t = 0; t < 16; ++t) if (il[t] == ibest) dl[t] = 3.4e38f;
        if (lane == 0) op[k] = ibest;
    }
}

// ---------------------------------------------------------------------------
// emb1: 128x9 CBR -> x0 point-major (B,N,128) f16
// ---------------------------------------------------------------------------
__global__ void emb1_kernel(const float* __restrict__ in, const float* __restrict__ w,
                            const float* __restrict__ s, const float* __restrict__ bi,
                            _Float16* __restrict__ x0) {
    int i = blockIdx.x * 256 + threadIdx.x;        // (b*NN+n)*CC + o
    int o = i % CC; int r = i / CC; int n = r % NN; int b = r / NN;
    const float* ib = in + (size_t)b * 9 * NN + n;
    float acc = 0.f;
#pragma unroll
    for (int c = 0; c < 9; ++c) acc += w[o * 9 + c] * ib[(size_t)c * NN];
    acc = fmaxf(s[o] * acc + bi[o], 0.f);
    x0[i] = (_Float16)acc;
}

// ---------------------------------------------------------------------------
// WMMA GEMM, point-major activations, Kc as compile-time constant so all
// fragment addresses fold into immediate offsets of two base pointers.
//   X: (B, Nn, Kc) f16   W: (Mo, Kc) f16   Y[b,n,o] = epilogue(sum_c W[o,c]X[b,n,c])
// One wave computes a 32(M) x 64(N) tile: 2x4 sub-tiles, 8 wmma/k-step with
// each weight fragment reused across 4 activation fragments.
// ---------------------------------------------------------------------------
struct GemmEp {
    const float* biasO;          // += biasO[o]               (pre-scale)
    const float* bias2d;         // += bias2d[b*Mo+o]         (pre-scale)
    const float* scale;          // y = scale[o]*y + shift[o]
    const float* shift;
    const float* resid;          // += resid[(b*Nn+n)*Mo+o]   (post-act), point-major
    int act;                     // 0 none, 1 relu, 2 leaky 0.2
    float*      out32;           // (B,Nn,Mo) point-major
    _Float16*   out16a; int a_ld; int a_coff;   // dst[(b*Nn+n)*a_ld + a_coff + o]
    _Float16*   out16b; int b_ld; int b_coff;
};

template <int Kc>
__global__ void gemm_wmma_kernel(const _Float16* __restrict__ W,
                                 const _Float16* __restrict__ X,
                                 int Mo, int Nn, GemmEp ep) {
    const int lane = threadIdx.x & 31;
    const int hi   = lane >> 4;
    const int lo   = lane & 15;
    const int n0 = blockIdx.x * 64;
    const int m0 = blockIdx.y * 32;
    const int b  = blockIdx.z;
    const _Float16* Xb = X + (size_t)b * Nn * Kc;

    // A: row m0(+16)+lo; B: point n0+ni*16+lo; per-lane 16B chunks, all
    // sub-tile strides (16*Kc halves) are compile-time immediates.
    const _Float16* ap = W  + (size_t)(m0 + lo) * Kc + hi * 16;
    const _Float16* bp = Xb + (size_t)(n0 + lo) * Kc + hi * 16;

    v8f acc[2][4] = {};
#pragma unroll 2
    for (int k0 = 0; k0 < Kc; k0 += 32) {
        union { v16h v; uint4 u[2]; } A0, A1, B0, B1, B2, B3;
        A0.u[0] = *(const uint4*)(ap);                A0.u[1] = *(const uint4*)(ap + 8);
        A1.u[0] = *(const uint4*)(ap + 16 * Kc);      A1.u[1] = *(const uint4*)(ap + 16 * Kc + 8);
        B0.u[0] = *(const uint4*)(bp);                B0.u[1] = *(const uint4*)(bp + 8);
        B1.u[0] = *(const uint4*)(bp + 16 * Kc);      B1.u[1] = *(const uint4*)(bp + 16 * Kc + 8);
        B2.u[0] = *(const uint4*)(bp + 32 * Kc);      B2.u[1] = *(const uint4*)(bp + 32 * Kc + 8);
        B3.u[0] = *(const uint4*)(bp + 48 * Kc);      B3.u[1] = *(const uint4*)(bp + 48 * Kc + 8);
        __builtin_prefetch(bp + 64, 0, 3);            // 2 k-steps ahead on the stream side
        __builtin_prefetch(bp + 16 * Kc + 64, 0, 3);
        __builtin_prefetch(bp + 32 * Kc + 64, 0, 3);
        __builtin_prefetch(bp + 48 * Kc + 64, 0, 3);
        acc[0][0] = __builtin_amdgcn_wmma_f32_16x16x32_f16(false, A0.v, false, B0.v, (short)0, acc[0][0], false, false);
        acc[0][1] = __builtin_amdgcn_wmma_f32_16x16x32_f16(false, A0.v, false, B1.v, (short)0, acc[0][1], false, false);
        acc[0][2] = __builtin_amdgcn_wmma_f32_16x16x32_f16(false, A0.v, false, B2.v, (short)0, acc[0][2], false, false);
        acc[0][3] = __builtin_amdgcn_wmma_f32_16x16x32_f16(false, A0.v, false, B3.v, (short)0, acc[0][3], false, false);
        acc[1][0] = __builtin_amdgcn_wmma_f32_16x16x32_f16(false, A1.v, false, B0.v, (short)0, acc[1][0], false, false);
        acc[1][1] = __builtin_amdgcn_wmma_f32_16x16x32_f16(false, A1.v, false, B1.v, (short)0, acc[1][1], false, false);
        acc[1][2] = __builtin_amdgcn_wmma_f32_16x16x32_f16(false, A1.v, false, B2.v, (short)0, acc[1][2], false, false);
        acc[1][3] = __builtin_amdgcn_wmma_f32_16x16x32_f16(false, A1.v, false, B3.v, (short)0, acc[1][3], false, false);
        ap += 32; bp += 32;
    }

#pragma unroll
    for (int mi = 0; mi < 2; ++mi) {
        const int obase = m0 + mi * 16 + hi * 8;   // this lane's 8 consecutive channels
        float sc[8], sh[8], bo[8], b2[8];
#pragma unroll
        for (int r = 0; r < 8; ++r) {
            sc[r] = ep.scale  ? ep.scale[obase + r]                      : 1.f;
            sh[r] = ep.shift  ? ep.shift[obase + r]                      : 0.f;
            bo[r] = ep.biasO  ? ep.biasO[obase + r]                      : 0.f;
            b2[r] = ep.bias2d ? ep.bias2d[(size_t)b * Mo + obase + r]    : 0.f;
        }
#pragma unroll
        for (int ni = 0; ni < 4; ++ni) {
            const int n = n0 + ni * 16 + lo;
            const v8f a = acc[mi][ni];
            float y[8];
#pragma unroll
            for (int r = 0; r < 8; ++r) {
                float v = a[r] + bo[r] + b2[r];
                v = sc[r] * v + sh[r];
                if (ep.act == 1) v = fmaxf(v, 0.f);
                else if (ep.act == 2) v = (v > 0.f) ? v : 0.2f * v;
                y[r] = v;
            }
            if (ep.resid) {
                const float4* rp = (const float4*)(ep.resid + ((size_t)b * Nn + n) * Mo + obase);
                float4 r0 = rp[0], r1 = rp[1];
                y[0] += r0.x; y[1] += r0.y; y[2] += r0.z; y[3] += r0.w;
                y[4] += r1.x; y[5] += r1.y; y[6] += r1.z; y[7] += r1.w;
            }
            if (ep.out32) {
                float4* wp = (float4*)(ep.out32 + ((size_t)b * Nn + n) * Mo + obase);
                float4 w0{y[0], y[1], y[2], y[3]}, w1{y[4], y[5], y[6], y[7]};
                wp[0] = w0; wp[1] = w1;
            }
            if (ep.out16a) {
                union { uint4 u; _Float16 h[8]; } st;
#pragma unroll
                for (int r = 0; r < 8; ++r) st.h[r] = (_Float16)y[r];
                *(uint4*)(ep.out16a + (size_t)((size_t)b * Nn + n) * ep.a_ld + ep.a_coff + obase) = st.u;
            }
            if (ep.out16b) {
                union { uint4 u; _Float16 h[8]; } st;
#pragma unroll
                for (int r = 0; r < 8; ++r) st.h[r] = (_Float16)y[r];
                *(uint4*)(ep.out16b + (size_t)((size_t)b * Nn + n) * ep.b_ld + ep.b_coff + obase) = st.u;
            }
        }
    }
}

// ---------------------------------------------------------------------------
// Per-point neighborhood attention: one wave per point, 4 channels per lane.
// q,k,v: (B,N,128) f32; agg: (B,N,128) f16 point-major (8B store per lane).
// ---------------------------------------------------------------------------
__global__ void attn_kernel(const float* __restrict__ q, const float* __restrict__ kk,
                            const float* __restrict__ vv, const float* __restrict__ in,
                            const int* __restrict__ idx, const float* __restrict__ wpos,
                            _Float16* __restrict__ agg) {
    const int lane = threadIdx.x & 31;
    const int wv   = threadIdx.x >> 5;
    const int p    = blockIdx.x * 8 + wv;
    const int b = p / NN, n = p % NN;
    const float* xb = in + (size_t)b * 9 * NN;
    const float qx = xb[n], qy = xb[NN + n], qz = xb[2 * NN + n];
    const float* qp = q + ((size_t)b * NN + n) * CC + lane * 4;
    const float q0 = qp[0], q1 = qp[1], q2 = qp[2], q3 = qp[3];
    const int* ip = idx + ((size_t)b * NN + n) * 16;

    float lg[16];
#pragma unroll
    for (int k = 0; k < 16; ++k) {
        int j = ip[k];
        const float* kp = kk + ((size_t)b * NN + j) * CC + lane * 4;
        float d = q0 * kp[0] + q1 * kp[1] + q2 * kp[2] + q3 * kp[3];
#pragma unroll
        for (int off = 16; off; off >>= 1) d += __shfl_xor(d, off, 32);
        lg[k] = d * 0.08838834764831845f;           // 1/sqrt(128)
    }
    float mx = lg[0];
#pragma unroll
    for (int k = 1; k < 16; ++k) mx = fmaxf(mx, lg[k]);
    float sum = 0.f;
#pragma unroll
    for (int k = 0; k < 16; ++k) { lg[k] = __expf(lg[k] - mx); sum += lg[k]; }
    const float inv = 1.f / sum;

    const int c0 = lane * 4;
    const float* wp = wpos + c0 * 3;
    const float w00 = wp[0], w01 = wp[1],  w02 = wp[2];
    const float w10 = wp[3], w11 = wp[4],  w12 = wp[5];
    const float w20 = wp[6], w21 = wp[7],  w22 = wp[8];
    const float w30 = wp[9], w31 = wp[10], w32 = wp[11];

    float a0 = 0.f, a1 = 0.f, a2 = 0.f, a3 = 0.f;
#pragma unroll
    for (int k = 0; k < 16; ++k) {
        int j = ip[k];
        float wgt = lg[k] * inv;
        float rx = qx - xb[j], ry = qy - xb[NN + j], rz = qz - xb[2 * NN + j];
        const float* vp = vv + ((size_t)b * NN + j) * CC + c0;
        a0 += wgt * (vp[0] + w00 * rx + w01 * ry + w02 * rz);
        a1 += wgt * (vp[1] + w10 * rx + w11 * ry + w12 * rz);
        a2 += wgt * (vp[2] + w20 * rx + w21 * ry + w22 * rz);
        a3 += wgt * (vp[3] + w30 * rx + w31 * ry + w32 * rz);
    }
    union { uint2 u; _Float16 h[4]; } st;
    st.h[0] = (_Float16)a0; st.h[1] = (_Float16)a1;
    st.h[2] = (_Float16)a2; st.h[3] = (_Float16)a3;
    *(uint2*)(agg + ((size_t)b * NN + n) * CC + c0) = st.u;
}

// ---------------------------------------------------------------------------
// g = concat(max_n, mean_n) of fuse output (B,N,1024 point-major).
// ---------------------------------------------------------------------------
__global__ void greduce_kernel(const _Float16* __restrict__ f, float* __restrict__ g) {
    int c = blockIdx.x % 1024, b = blockIdx.x / 1024;
    const _Float16* src = f + (size_t)b * NN * 1024 + c;
    float mx = -3.4e38f, sm = 0.f;
    for (int n = threadIdx.x; n < NN; n += 256) {
        float v = (float)src[(size_t)n * 1024];
        mx = fmaxf(mx, v); sm += v;
    }
    __shared__ float smx[256], ssm[256];
    smx[threadIdx.x] = mx; ssm[threadIdx.x] = sm;
    __syncthreads();
    for (int s = 128; s; s >>= 1) {
        if ((int)threadIdx.x < s) {
            smx[threadIdx.x] = fmaxf(smx[threadIdx.x], smx[threadIdx.x + s]);
            ssm[threadIdx.x] += ssm[threadIdx.x + s];
        }
        __syncthreads();
    }
    if (threadIdx.x == 0) {
        g[(size_t)b * 2048 + c]        = smx[0];
        g[(size_t)b * 2048 + 1024 + c] = ssm[0] * (1.f / NN);
    }
}

// bias_g[b,o] = cls_bias1[o] + sum_c cls_w1[o,1024+c] * g[b,c]
__global__ void biasg_kernel(const float* __restrict__ w1, const float* __restrict__ bias1,
                             const float* __restrict__ g, float* __restrict__ bg) {
    int i = blockIdx.x * 256 + threadIdx.x;         // B*512
    if (i >= BB * 512) return;
    int o = i % 512, b = i / 512;
    const float* gb = g + (size_t)b * 2048;
    const float* wr = w1 + (size_t)o * 3072 + 1024;
    float acc = bias1[o];
    for (int c = 0; c < 2048; ++c) acc += wr[c] * gb[c];
    bg[i] = acc;
}

// cls3: 13x256 + bias; x point-major (B,N,256) contiguous reads; out (B,13,N)
__global__ void cls3_kernel(const _Float16* __restrict__ x, const float* __restrict__ w,
                            const float* __restrict__ bias, float* __restrict__ out) {
    int i = blockIdx.x * 256 + threadIdx.x;         // B*13*N
    if (i >= BB * 13 * NN) return;
    int n = i % NN; int r = i / NN; int o = r % 13; int b = r / 13;
    const _Float16* xb = x + ((size_t)b * NN + n) * 256;
    float acc = bias[o];
    for (int c = 0; c < 256; ++c) acc += w[o * 256 + c] * (float)xb[c];
    out[i] = acc;
}

// ---------------------------------------------------------------------------
extern "C" void kernel_launch(void* const* d_in, const int* in_sizes, int n_in,
                              void* d_out, int out_size, void* d_ws, size_t ws_size,
                              hipStream_t stream) {
    const float* inputs   = (const float*)d_in[0];
    const float* emb_w1   = (const float*)d_in[1];
    const float* emb_s1   = (const float*)d_in[2];
    const float* emb_b1   = (const float*)d_in[3];
    const float* emb_w2   = (const float*)d_in[4];
    const float* emb_s2   = (const float*)d_in[5];
    const float* emb_b2   = (const float*)d_in[6];
    const float* blk_wq   = (const float*)d_in[7];
    const float* blk_wk   = (const float*)d_in[8];
    const float* blk_wv   = (const float*)d_in[9];
    const float* blk_wpos = (const float*)d_in[10];
    const float* blk_wo   = (const float*)d_in[11];
    const float* blk_s    = (const float*)d_in[12];
    const float* blk_b    = (const float*)d_in[13];
    const float* fuse_w   = (const float*)d_in[14];
    const float* fuse_s   = (const float*)d_in[15];
    const float* fuse_b   = (const float*)d_in[16];
    const float* cls_w1   = (const float*)d_in[17];
    const float* cls_b1   = (const float*)d_in[18];
    const float* cls_s1   = (const float*)d_in[19];
    const float* cls_sh1  = (const float*)d_in[20];
    const float* cls_w2   = (const float*)d_in[21];
    const float* cls_b2   = (const float*)d_in[22];
    const float* cls_s2   = (const float*)d_in[23];
    const float* cls_sh2  = (const float*)d_in[24];
    const float* cls_w3   = (const float*)d_in[25];
    const float* cls_b3   = (const float*)d_in[26];
    float* out = (float*)d_out;

    // workspace carve-out
    char* p = (char*)d_ws;
    auto alloc = [&](size_t bytes) -> void* {
        void* r = (void*)p; p += (bytes + 255) & ~(size_t)255; return r;
    };
    int*       idx     = (int*)      alloc((size_t)BB * NN * 16 * 4);
    float*     x32     = (float*)    alloc((size_t)BB * NN * CC * 4);   // (B,N,128)
    _Float16*  x16     = (_Float16*) alloc((size_t)BB * NN * CC * 2);
    _Float16*  x016    = (_Float16*) alloc((size_t)BB * NN * CC * 2);
    _Float16*  xs16    = (_Float16*) alloc((size_t)BB * NN * 3 * CC * 2); // (B,N,384)
    float*     q32     = (float*)    alloc((size_t)BB * NN * CC * 4);
    float*     k32     = (float*)    alloc((size_t)BB * NN * CC * 4);
    float*     v32     = (float*)    alloc((size_t)BB * NN * CC * 4);
    _Float16*  agg16   = (_Float16*) alloc((size_t)BB * NN * CC * 2);
    _Float16*  fuse16o = (_Float16*) alloc((size_t)BB * NN * 1024 * 2); // (B,N,1024)
    float*     gvec    = (float*)    alloc((size_t)BB * 2048 * 4);
    float*     bg      = (float*)    alloc((size_t)BB * 512 * 4);
    _Float16*  c1o16   = (_Float16*) alloc((size_t)BB * NN * 512 * 2);
    _Float16*  c2o16   = (_Float16*) alloc((size_t)BB * NN * 256 * 2);
    _Float16*  embw2h  = (_Float16*) alloc((size_t)CC * CC * 2);
    _Float16*  wqh     = (_Float16*) alloc((size_t)3 * CC * CC * 2);
    _Float16*  wkh     = (_Float16*) alloc((size_t)3 * CC * CC * 2);
    _Float16*  wvh     = (_Float16*) alloc((size_t)3 * CC * CC * 2);
    _Float16*  woh     = (_Float16*) alloc((size_t)3 * CC * CC * 2);
    _Float16*  fuseh   = (_Float16*) alloc((size_t)1024 * 384 * 2);
    _Float16*  w1ah    = (_Float16*) alloc((size_t)512 * 1024 * 2);
    _Float16*  w2h     = (_Float16*) alloc((size_t)256 * 512 * 2);

    auto cvt = [&](const float* s, _Float16* d, int n) {
        cvt_f16_kernel<<<(n + 255) / 256, 256, 0, stream>>>(s, d, n);
    };
    cvt(emb_w2, embw2h, CC * CC);
    cvt(blk_wq, wqh, 3 * CC * CC);
    cvt(blk_wk, wkh, 3 * CC * CC);
    cvt(blk_wv, wvh, 3 * CC * CC);
    cvt(blk_wo, woh, 3 * CC * CC);
    cvt(fuse_w, fuseh, 1024 * 384);
    cvt(cls_w2, w2h, 256 * 512);
    cvt_w1a_kernel<<<(512 * 1024) / 256, 256, 0, stream>>>(cls_w1, w1ah);

    knn_kernel<<<(BB * NN) / 8, 256, 0, stream>>>(inputs, idx);
    emb1_kernel<<<(BB * NN * CC) / 256, 256, 0, stream>>>(inputs, emb_w1, emb_s1, emb_b1, x016);

    auto gemm = [&](const _Float16* W, const _Float16* X, int Mo, int Kc, GemmEp ep) {
        dim3 grid(NN / 64, Mo / 32, BB);
        switch (Kc) {
            case 128:  gemm_wmma_kernel<128> <<<grid, 32, 0, stream>>>(W, X, Mo, NN, ep); break;
            case 384:  gemm_wmma_kernel<384> <<<grid, 32, 0, stream>>>(W, X, Mo, NN, ep); break;
            case 512:  gemm_wmma_kernel<512> <<<grid, 32, 0, stream>>>(W, X, Mo, NN, ep); break;
            case 1024: gemm_wmma_kernel<1024><<<grid, 32, 0, stream>>>(W, X, Mo, NN, ep); break;
        }
    };

    {   // emb2: relu(s2*(W x)+b2) -> x32 / x16
        GemmEp ep{}; ep.scale = emb_s2; ep.shift = emb_b2; ep.act = 1;
        ep.out32 = x32; ep.out16a = x16; ep.a_ld = CC; ep.a_coff = 0;
        gemm(embw2h, x016, CC, CC, ep);
    }

    for (int i = 0; i < 3; ++i) {
        GemmEp eq{}; eq.out32 = q32; gemm(wqh + (size_t)i * CC * CC, x16, CC, CC, eq);
        GemmEp ek{}; ek.out32 = k32; gemm(wkh + (size_t)i * CC * CC, x16, CC, CC, ek);
        GemmEp ev{}; ev.out32 = v32; gemm(wvh + (size_t)i * CC * CC, x16, CC, CC, ev);

        attn_kernel<<<(BB * NN) / 8, 256, 0, stream>>>(q32, k32, v32, inputs, idx,
                                                       blk_wpos + (size_t)i * CC * 3, agg16);

        GemmEp eo{};                    // x = x + relu(s*Wo agg + b); also xs slice
        eo.scale = blk_s + (size_t)i * CC; eo.shift = blk_b + (size_t)i * CC; eo.act = 1;
        eo.resid = x32; eo.out32 = x32;
        eo.out16a = x16;  eo.a_ld = CC;     eo.a_coff = 0;
        eo.out16b = xs16; eo.b_ld = 3 * CC; eo.b_coff = i * CC;
        gemm(woh + (size_t)i * CC * CC, agg16, CC, CC, eo);
    }

    {   // fuse: leaky(s*(W xs)+b) -> fuse16o (B,N,1024)
        GemmEp ep{}; ep.scale = fuse_s; ep.shift = fuse_b; ep.act = 2;
        ep.out16a = fuse16o; ep.a_ld = 1024;
        gemm(fuseh, xs16, 1024, 3 * CC, ep);
    }

    greduce_kernel<<<BB * 1024, 256, 0, stream>>>(fuse16o, gvec);
    biasg_kernel<<<(BB * 512 + 255) / 256, 256, 0, stream>>>(cls_w1, cls_b1, gvec, bg);

    {   // cls1: relu(s1*(W1a x + bias1 + W1b g) + sh1)
        GemmEp ep{}; ep.bias2d = bg; ep.scale = cls_s1; ep.shift = cls_sh1; ep.act = 1;
        ep.out16a = c1o16; ep.a_ld = 512;
        gemm(w1ah, fuse16o, 512, 1024, ep);
    }
    {   // cls2: relu(s2*(W2 x + bias2) + sh2)
        GemmEp ep{}; ep.biasO = cls_b2; ep.scale = cls_s2; ep.shift = cls_sh2; ep.act = 1;
        ep.out16a = c2o16; ep.a_ld = 256;
        gemm(w2h, c1o16, 256, 512, ep);
    }

    cls3_kernel<<<(BB * 13 * NN + 255) / 256, 256, 0, stream>>>(c2o16, cls_w3, cls_b3, out);

    (void)in_sizes; (void)n_in; (void)out_size; (void)ws_size;
}
